// sigma_pi_6502580486513
// MI455X (gfx1250) — matrix operand order repaired and checked by reading, not compiled, not run on hardware
//
#include <hip/hip_runtime.h>
#include <stdint.h>

// Problem constants (from the reference)
constexpr int Bn = 2048, Dn = 64, Mn = 4, Nn = 64, Cn = 256;

// Tiling: each wave owns BT 16-row b-tiles x one 16-col c-tile.
constexpr int BT = 2;                    // b-tiles per wave (keeps VGPRs < 256)
constexpr int WAVES = 4;                 // waves per block (wave32 -> 128 threads)
constexpr int THREADS = WAVES * 32;
constexpr int ROWS_PER_BLOCK = WAVES * BT * 16;  // 128 batch rows per block

// LDS staging of the per-n B panel: vs[:, :, n, c0:c0+16] stored transposed as
// [m][c][d] with the d-row padded 64->68 floats (272 B) so the 16 column-lanes
// of a ds_load_b128 land on disjoint bank quads (4L..4L+3 mod 64).
constexpr int DPAD = 68;
constexpr int SLOT = Mn * 16 * DPAD;     // floats per buffer (4352)

typedef __attribute__((ext_vector_type(16))) __bf16 v16bf;
typedef __attribute__((ext_vector_type(8)))  float  v8f;

__device__ __forceinline__ float relu1(float x) {
    // Single v_med3_num_f32: med3(x, 0, +inf) == max(x, 0)
    return __builtin_amdgcn_fmed3f(x, 0.0f, __builtin_inff());
}

__global__ __launch_bounds__(THREADS)
void sigma_pi_wmma_kernel(const float* __restrict__ xb,
                          const float* __restrict__ vs,
                          const float* __restrict__ alphas,
                          const float* __restrict__ betas,
                          const float* __restrict__ bias,
                          const float* __restrict__ ws,
                          float* __restrict__ out) {
    __shared__ float smem[2][SLOT];

    const int tid  = threadIdx.x;
    const int lane = tid & 31;
    const int wave = tid >> 5;
    const int hi   = lane >> 4;          // half-wave select (0: lanes 0-15, 1: 16-31)
    const int lrow = lane & 15;
    const int c0   = blockIdx.x * 16;
    const int cidx = c0 + lrow;          // this lane's output column
    const int row_base = blockIdx.y * ROWS_PER_BLOCK + wave * (BT * 16);

    // 32-bit LDS byte address (flat shared ptr low bits = LDS offset).
    const unsigned lds_base = (unsigned)(uintptr_t)(&smem[0][0]);

    // Per-thread constant part of the staging mapping:
    //   flat = q*128 + tid;  i = tid&15 (c);  dm = 8q + (tid>>4) = d*4+m
    const int t4  = tid >> 4;            // 0..7
    const int s_i = tid & 15;            // c within tile (constant per thread)
    const int s_m = t4 & 3;              // m (constant per thread)
    const size_t GSTEP = (size_t)8 * Nn * Cn;        // dm += 8 per step (floats)

    // Async-DMA one n-panel (64d x 4m x 16c floats = 16 KB) into smem[buf],
    // transposing to [m][c][d+pad]. Addresses advance linearly per step.
    auto stage = [&](int n, int buf) {
        const float* g = vs + (size_t)(t4 * Nn + n) * Cn + c0 + s_i;
        unsigned l = lds_base +
            4u * (unsigned)(buf * SLOT + (s_m * 16 + s_i) * DPAD + (t4 >> 2));
#pragma unroll 4
        for (int q = 0; q < 32; ++q) {
            asm volatile("global_load_async_to_lds_b32 %0, %1, off"
                         :: "v"(l), "v"(g) : "memory");
            g += GSTEP;   // next dm group of 8
            l += 8u;      // d += 2 floats
        }
    };

    // Per-n unit scalars (depend only on this lane's column).
    auto load_scalars = [&](int n, float* al, float* be, float* bi, float& wsv) {
#pragma unroll
        for (int m = 0; m < Mn; ++m) {
            const int unc = m * (Nn * Cn) + n * Cn + cidx;
            al[m] = alphas[unc];
            be[m] = betas[unc];
            bi[m] = bias[unc];
        }
        wsv = ws[n * Cn + cidx];
    };

    // ---- Load the 16x64 xb panel per b-tile, convert f32->bf16 into WMMA A
    //      layout (16-bit A 16x32: per lane K = hi*8+[0..7] in elems 0-7,
    //      K = 16+hi*8+[0..7] in elems 8-15). Two contiguous float4 pairs. ----
    v16bf a[BT][2];
#pragma unroll
    for (int bt = 0; bt < BT; ++bt) {
#pragma unroll
        for (int kc = 0; kc < 2; ++kc) {
            const float* src = xb + (size_t)(row_base + bt * 16 + lrow) * Dn
                                  + kc * 32 + hi * 8;
            const float4* p = reinterpret_cast<const float4*>(src);
            float4 x0 = p[0], x1 = p[1];   // K = base .. base+7
            float4 x2 = p[4], x3 = p[5];   // K = base+16 .. base+23
            v16bf av;
            av[0]  = (__bf16)x0.x; av[1]  = (__bf16)x0.y;
            av[2]  = (__bf16)x0.z; av[3]  = (__bf16)x0.w;
            av[4]  = (__bf16)x1.x; av[5]  = (__bf16)x1.y;
            av[6]  = (__bf16)x1.z; av[7]  = (__bf16)x1.w;
            av[8]  = (__bf16)x2.x; av[9]  = (__bf16)x2.y;
            av[10] = (__bf16)x2.z; av[11] = (__bf16)x2.w;
            av[12] = (__bf16)x3.x; av[13] = (__bf16)x3.y;
            av[14] = (__bf16)x3.z; av[15] = (__bf16)x3.w;
            a[bt][kc] = av;
        }
    }

    v8f outacc[BT];
#pragma unroll
    for (int bt = 0; bt < BT; ++bt)
#pragma unroll
        for (int e = 0; e < 8; ++e) outacc[bt][e] = 0.0f;

    // Prologue: stage panel n=0 + scalars for n=0; wait; make visible.
    float al[Mn], be[Mn], bi[Mn], wsv;
    stage(0, 0);
    load_scalars(0, al, be, bi, wsv);
    asm volatile("s_wait_asynccnt 0x0" ::: "memory");
    __syncthreads();

    for (int n = 0; n < Nn; ++n) {
        const int buf = n & 1;

        // Overlap next panel's DMA + scalar loads with this panel's WMMAs.
        float aln[Mn], ben[Mn], bin[Mn], wsn = 0.0f;
        if (n + 1 < Nn) {
            stage(n + 1, buf ^ 1);
            load_scalars(n + 1, aln, ben, bin, wsn);
        }

        v8f pacc[BT];                             // Pi over m, per b-tile
#pragma unroll
        for (int bt = 0; bt < BT; ++bt)
#pragma unroll
            for (int e = 0; e < 8; ++e) pacc[bt][e] = 1.0f;

#pragma unroll
        for (int m = 0; m < Mn; ++m) {
            // B-operand from LDS: per lane column lrow, K = kc*32 + hi*8 + [0..7]
            // and kc*32 + 16 + hi*8 + [0..7] -> 4x ds_load_b128 each chunk, then pack.
            const float* sb = &smem[buf][(m * 16 + lrow) * DPAD + hi * 8];
            const float4* p4 = reinterpret_cast<const float4*>(sb);
            float4 y0 = p4[0], y1 = p4[1], y2 = p4[4], y3 = p4[5];     // kc=0
            float4 y4 = p4[8], y5 = p4[9], y6 = p4[12], y7 = p4[13];   // kc=1

            v16bf b0, b1;
            b0[0]  = (__bf16)y0.x; b0[1]  = (__bf16)y0.y;
            b0[2]  = (__bf16)y0.z; b0[3]  = (__bf16)y0.w;
            b0[4]  = (__bf16)y1.x; b0[5]  = (__bf16)y1.y;
            b0[6]  = (__bf16)y1.z; b0[7]  = (__bf16)y1.w;
            b0[8]  = (__bf16)y2.x; b0[9]  = (__bf16)y2.y;
            b0[10] = (__bf16)y2.z; b0[11] = (__bf16)y2.w;
            b0[12] = (__bf16)y3.x; b0[13] = (__bf16)y3.y;
            b0[14] = (__bf16)y3.z; b0[15] = (__bf16)y3.w;
            b1[0]  = (__bf16)y4.x; b1[1]  = (__bf16)y4.y;
            b1[2]  = (__bf16)y4.z; b1[3]  = (__bf16)y4.w;
            b1[4]  = (__bf16)y5.x; b1[5]  = (__bf16)y5.y;
            b1[6]  = (__bf16)y5.z; b1[7]  = (__bf16)y5.w;
            b1[8]  = (__bf16)y6.x; b1[9]  = (__bf16)y6.y;
            b1[10] = (__bf16)y6.z; b1[11] = (__bf16)y6.w;
            b1[12] = (__bf16)y7.x; b1[13] = (__bf16)y7.y;
            b1[14] = (__bf16)y7.z; b1[15] = (__bf16)y7.w;

            // bias[m,n,c] depends only on the column -> splat as the C seed.
            v8f cb;
#pragma unroll
            for (int e = 0; e < 8; ++e) cb[e] = bi[m];

#pragma unroll
            for (int bt = 0; bt < BT; ++bt) {
                // z = xb_tile @ vs_slice + bias (16x16x32 bf16 WMMA, f32 accum)
                v8f z = __builtin_amdgcn_wmma_f32_16x16x32_bf16(
                    false, a[bt][0], false, b0, (short)0, cb, false, false);
                z = __builtin_amdgcn_wmma_f32_16x16x32_bf16(
                    false, a[bt][1], false, b1, (short)0, z, false, false);

                // t = alpha + beta * relu(z); pacc *= t (3 VALU / element)
#pragma unroll
                for (int e = 0; e < 8; ++e) {
                    float t = __builtin_fmaf(be[m], relu1(z[e]), al[m]);
                    pacc[bt][e] *= t;
                }
            }
        }

#pragma unroll
        for (int bt = 0; bt < BT; ++bt)
#pragma unroll
            for (int e = 0; e < 8; ++e)
                outacc[bt][e] += pacc[bt][e] * wsv;

        // Rotate pipelined scalars into place.
        if (n + 1 < Nn) {
#pragma unroll
            for (int m = 0; m < Mn; ++m) { al[m] = aln[m]; be[m] = ben[m]; bi[m] = bin[m]; }
            wsv = wsn;
        }

        // Next panel's DMA must be complete & visible before consumption;
        // barrier also fences buffer reuse two iterations out.
        asm volatile("s_wait_asynccnt 0x0" ::: "memory");
        __syncthreads();
    }

    // Store: D-layout row M = vgpr + hi*8, column = lane&15.
#pragma unroll
    for (int bt = 0; bt < BT; ++bt)
#pragma unroll
        for (int r = 0; r < 8; ++r)
            out[(size_t)(row_base + bt * 16 + hi * 8 + r) * Cn + cidx] = outacc[bt][r];
}

extern "C" void kernel_launch(void* const* d_in, const int* in_sizes, int n_in,
                              void* d_out, int out_size, void* d_ws, size_t ws_size,
                              hipStream_t stream) {
    const float* xb     = (const float*)d_in[0];
    const float* vs     = (const float*)d_in[1];
    const float* alphas = (const float*)d_in[2];
    const float* betas  = (const float*)d_in[3];
    const float* bias   = (const float*)d_in[4];
    const float* ws     = (const float*)d_in[5];
    float* out = (float*)d_out;

    dim3 grid(Cn / 16, Bn / ROWS_PER_BLOCK);   // (16, 16)
    dim3 block(THREADS);                       // 128 threads = 4 wave32
    hipLaunchKernelGGL(sigma_pi_wmma_kernel, grid, block, 0, stream,
                       xb, vs, alphas, betas, bias, ws, out);
}
